// Learnable_Topology_GAT_4011499454474
// MI455X (gfx1250) — matrix operand order repaired and checked by reading, not compiled, not run on hardware
//
#include <hip/hip_runtime.h>
#include <hip/hip_bf16.h>

typedef __attribute__((ext_vector_type(16))) _Float16 v16h;
typedef __attribute__((ext_vector_type(8)))  float    v8f;

union HV16 { v16h v; uint4 u[2]; };

#define NEG_INF_F (-9.0e15f)
#define FR_LD 40   // padded LDS stride (halves): 80B, multiple of 16B

// Issue an async GL2->LDS 16B copy (gfx1250 GLOBAL_LOAD_ASYNC_TO_LDS_B128, GV mode).
// ldsPtr: generic pointer into __shared__ (low 32 bits == LDS byte offset).
__device__ __forceinline__ void async_copy_b128(const void* gptr, void* ldsPtr)
{
    unsigned lds = (unsigned)(size_t)ldsPtr;
    unsigned long long g = (unsigned long long)(size_t)gptr;
    asm volatile("global_load_async_to_lds_b128 %0, %1, off"
                 :: "v"(lds), "v"(g) : "memory");
}
__device__ __forceinline__ void wait_async0()
{
    asm volatile("s_wait_asynccnt 0x0" ::: "memory");
}

// ---------------------------------------------------------------------------
// WMMA GEMM: C[M,N] = act( A_f16[M,K] @ B + bias ) (* rowscale)
// B given TRANSPOSED (K-major panels): BT[n*ldbt + k].
// Double-buffered LDS, filled by ASYNC global->LDS copies (ASYNCcnt), tiles in
// *fragment order* so each operand fragment is 2x ds_load_b128 per lane:
//   A-frag slot s (=2v+q) of lane (m,hi) holds A[m][k], k = hi*8 + s + (s>=8?8:0)
//     -> chunk c8 lands at  As[row*FR_LD + ((c8>>3)&1)*16 + ((c8>>4)<<3)]
//   B-frag slot s of lane (n,hi) holds B[k][n],  k = hi*8 + s + (s>=8?8:0)
//     -> chunk c8 lands at  Bs[col*FR_LD + ((c8>>3)&1)*16 + ((c8>>4)<<3)]
// ---------------------------------------------------------------------------
__global__ void __launch_bounds__(256)
wmma_gemm(const _Float16* __restrict__ A, const _Float16* __restrict__ BT,
          float* __restrict__ C,
          int M, int N, int K, int lda, int ldbt, int ldc,
          const float* __restrict__ bias, const float* __restrict__ rowscale,
          int act, long sA, long sB, long sC)
{
    A  += (size_t)blockIdx.z * sA;
    BT += (size_t)blockIdx.z * sB;
    C  += (size_t)blockIdx.z * sC;

    __shared__ __align__(16) _Float16 AsF[2][64 * FR_LD];
    __shared__ __align__(16) _Float16 BsF[2][64 * FR_LD];

    const int tid  = threadIdx.x;
    const int wave = tid >> 5, lane = tid & 31;
    const int wr = wave >> 1, wc = wave & 1;
    const int m0 = blockIdx.y * 64, n0 = blockIdx.x * 64;
    const int hi = lane >> 4, l16 = lane & 15;

    // staging roles (one 8-half chunk each for A and B)
    const int sr  = tid >> 2;              // 0..63 : A row / B col in tile
    const int c8  = (tid & 3) << 3;        // 0,8,16,24
    const _Float16* Ag = A  + (size_t)(m0 + sr) * lda  + c8;
    const _Float16* Bg = BT + (size_t)(n0 + sr) * ldbt + c8;
    const int dstA = sr * FR_LD + ((c8 >> 3) & 1) * 16 + ((c8 >> 4) << 3);
    const int dstB = sr * FR_LD + ((c8 >> 3) & 1) * 16 + ((c8 >> 4) << 3);

    const int aFragOff  = (wr * 16 + l16)      * FR_LD + hi * 16;
    const int bFragOff0 = (wc * 32 + l16)      * FR_LD + hi * 16;
    const int bFragOff1 = (wc * 32 + 16 + l16) * FR_LD + hi * 16;

    auto issue = [&](int k0, int buf) {
        async_copy_b128(Ag + k0, &AsF[buf][dstA]);
        async_copy_b128(Bg + k0, &BsF[buf][dstB]);
    };

    v8f acc0 = {}, acc1 = {};

    issue(0, 0);
    int cur = 0;
    for (int k0 = 0; k0 < K; k0 += 32) {
        wait_async0();
        __syncthreads();                 // buf[cur] visible to all waves
        if (k0 + 32 < K) {
            issue(k0 + 32, cur ^ 1);     // overlap next chunk with compute
            if (k0 + 64 < K) {
                __builtin_prefetch(Ag + k0 + 64);
                __builtin_prefetch(Bg + k0 + 64);
            }
        }
        HV16 a, b0, b1;
        const uint4* ap  = (const uint4*)&AsF[cur][aFragOff];
        const uint4* bp0 = (const uint4*)&BsF[cur][bFragOff0];
        const uint4* bp1 = (const uint4*)&BsF[cur][bFragOff1];
        a.u[0]  = ap[0];  a.u[1]  = ap[1];
        b0.u[0] = bp0[0]; b0.u[1] = bp0[1];
        b1.u[0] = bp1[0]; b1.u[1] = bp1[1];
        acc0 = __builtin_amdgcn_wmma_f32_16x16x32_f16(false, a.v, false, b0.v, (short)0, acc0, false, false);
        acc1 = __builtin_amdgcn_wmma_f32_16x16x32_f16(false, a.v, false, b1.v, (short)0, acc1, false, false);
        cur ^= 1;
    }

    const int nc = n0 + wc * 32 + l16;
    #pragma unroll
    for (int v = 0; v < 8; v++) {
        int row = m0 + wr * 16 + v + (hi << 3);
        float v0 = acc0[v], v1 = acc1[v];
        if (bias)     { v0 += bias[nc]; v1 += bias[nc + 16]; }
        if (rowscale) { float rs = rowscale[row]; v0 *= rs; v1 *= rs; }
        if (act == 1) { v0 = fmaxf(v0, 0.f); v1 = fmaxf(v1, 0.f); }
        C[(size_t)row * ldc + nc]      = v0;
        C[(size_t)row * ldc + nc + 16] = v1;
    }
}

// ---------------------------------------------------------------------------
// Elementwise / layout helpers
// ---------------------------------------------------------------------------
__global__ void conv_f16(const float* __restrict__ s, _Float16* __restrict__ d, size_t n)
{
    size_t i = (size_t)blockIdx.x * blockDim.x + threadIdx.x;
    if (i < n) d[i] = (_Float16)s[i];
}

// out[b][c][r] = (f16) in[b][r][c]   (K-major B panels for WMMA)
__global__ void tconv_f16(const float* __restrict__ in, _Float16* __restrict__ out,
                          int R, int C, size_t total)
{
    size_t i = (size_t)blockIdx.x * blockDim.x + threadIdx.x;
    if (i >= total) return;
    size_t rc = (size_t)R * C;
    size_t b = i / rc, rem = i - b * rc;
    int r = (int)(rem / C), c = (int)(rem - (size_t)r * C);
    out[b * rc + (size_t)c * R + r] = (_Float16)in[i];
}

__global__ void relu_conv(const float* __restrict__ s, float* __restrict__ d,
                          _Float16* __restrict__ dh, size_t n)
{
    size_t i = (size_t)blockIdx.x * blockDim.x + threadIdx.x;
    if (i < n) { float v = fmaxf(s[i], 0.f); d[i] = v; dh[i] = (_Float16)v; }
}

__global__ void add_conv(const float* __restrict__ a, const float* __restrict__ b,
                         float* __restrict__ o, _Float16* __restrict__ oh, size_t n)
{
    size_t i = (size_t)blockIdx.x * blockDim.x + threadIdx.x;
    if (i < n) { float v = a[i] + b[i]; o[i] = v; oh[i] = (_Float16)v; }
}

__global__ void x3_elu(const float* __restrict__ g, const float* __restrict__ x2,
                       _Float16* __restrict__ x3h, size_t n)
{
    size_t i = (size_t)blockIdx.x * blockDim.x + threadIdx.x;
    if (i < n) {
        float v = g[i] + x2[(size_t)3072 * 256 + i];
        v = v > 0.f ? v : (__expf(v) - 1.f);
        x3h[i] = (_Float16)v;
    }
}

// ---------------------------------------------------------------------------
// Adjacency pieces
// ---------------------------------------------------------------------------
__global__ void row_normalize(const float* __restrict__ af, float* __restrict__ naf,
                              _Float16* __restrict__ nafh, int N)
{
    int r = blockIdx.x * blockDim.x + threadIdx.x;
    if (r >= N) return;
    float ss = 0.f;
    for (int k = 0; k < 64; k++) { float v = af[(size_t)r * 64 + k]; ss += v * v; }
    float inv = 1.f / fmaxf(sqrtf(ss), 1e-12f);
    for (int k = 0; k < 64; k++) {
        float v = af[(size_t)r * 64 + k] * inv;
        naf[(size_t)r * 64 + k]  = v;
        nafh[(size_t)r * 64 + k] = (_Float16)v;
    }
}

__global__ void colsum64(const float* __restrict__ naf, float* __restrict__ s, int N)
{
    int c = blockIdx.x;
    __shared__ float red[256];
    float acc = 0.f;
    for (int r = threadIdx.x; r < N; r += 256) acc += naf[(size_t)r * 64 + c];
    red[threadIdx.x] = acc; __syncthreads();
    for (int st = 128; st > 0; st >>= 1) {
        if (threadIdx.x < st) red[threadIdx.x] += red[threadIdx.x + st];
        __syncthreads();
    }
    if (threadIdx.x == 0) s[c] = red[0];
}

__global__ void rinv_kernel(const float* __restrict__ naf, const float* __restrict__ s,
                            float* __restrict__ rinv, int N)
{
    int r = blockIdx.x * blockDim.x + threadIdx.x;
    if (r >= N) return;
    float d = 0.f;
    for (int k = 0; k < 64; k++) d += naf[(size_t)r * 64 + k] * s[k];
    float ri = 1.f / d;
    if (isinf(ri)) ri = 0.f;
    rinv[r] = ri;
}

// ---------------------------------------------------------------------------
// Attention logits f1/f2
// ---------------------------------------------------------------------------
__global__ void f12_mh(const float* __restrict__ Wh, const float* __restrict__ a,
                       float* __restrict__ f1, float* __restrict__ f2, int N)
{
    int idx = blockIdx.x * blockDim.x + threadIdx.x;
    if (idx >= 4 * N) return;
    int h = idx / N, n = idx - h * N;
    const float* w  = Wh + ((size_t)h * N + n) * 64;
    const float* av = a + h * 128;
    float s1 = 0.f, s2 = 0.f;
    for (int k = 0; k < 64; k++) { s1 += w[k] * av[k]; s2 += w[k] * av[64 + k]; }
    f1[idx] = s1; f2[idx] = s2;
}

__global__ void f12_sg(const float* __restrict__ Wh, const float* __restrict__ a,
                       float* __restrict__ f1, float* __restrict__ f2, int N)
{
    int n = blockIdx.x * blockDim.x + threadIdx.x;
    if (n >= N) return;
    const float* w = Wh + (size_t)n * 256;
    float s1 = 0.f, s2 = 0.f;
    for (int k = 0; k < 256; k++) { s1 += w[k] * a[k]; s2 += w[k] * a[256 + k]; }
    f1[n] = s1; f2[n] = s2;
}

__device__ __forceinline__ float gat_e(float adjv, float fi, float f2j)
{
    if (adjv > 0.f) { float t = fi + f2j; return t > 0.f ? t : 0.2f * t; }
    return NEG_INF_F;
}

// Build P-fragment inputs: one contiguous 16-float run per lane
__device__ __forceinline__ void load_runs(const float* __restrict__ p, int j0, int hi,
                                          float* __restrict__ dst /*alignas16 [16]*/)
{
    const float* q = p + j0 + hi * 16;
    *(float4*)&dst[0]  = *(const float4*)(q);
    *(float4*)&dst[4]  = *(const float4*)(q + 4);
    *(float4*)&dst[8]  = *(const float4*)(q + 8);
    *(float4*)&dst[12] = *(const float4*)(q + 12);
}

// ---------------------------------------------------------------------------
// Multi-head masked attention, flash-style, WMMA P@V. Block=128 (4 waves),
// 16-row tile per block, one 16-col d-tile per wave. Wht is [H][64][N] K-major.
// ---------------------------------------------------------------------------
__global__ void __launch_bounds__(128)
gat_mh_attn(const float* __restrict__ adj, const float* __restrict__ f1,
            const float* __restrict__ f2, const _Float16* __restrict__ Wht,
            float* __restrict__ feat, int N)
{
    const int h  = blockIdx.y;
    const int i0 = blockIdx.x * 16;
    const int tid = threadIdx.x, wave = tid >> 5, lane = tid & 31;
    const float* f1h = f1 + (size_t)h * N;
    const float* f2h = f2 + (size_t)h * N;
    const _Float16* Wpt = Wht + (size_t)h * 64 * N;

    __shared__ float red[16][8];
    __shared__ float rowm[16], rowinv[16];

    // ---- pass 1: per-row max, then sum (8 threads/row, float4 reads) ----
    int r = tid >> 3, jt = tid & 7;
    float fi = f1h[i0 + r];
    const float* arow = adj + (size_t)(i0 + r) * N;
    float mx = NEG_INF_F;
    for (int j = jt * 4; j < N; j += 32) {
        float4 a4 = *(const float4*)(arow + j);
        float4 g4 = *(const float4*)(f2h + j);
        mx = fmaxf(mx, gat_e(a4.x, fi, g4.x));
        mx = fmaxf(mx, gat_e(a4.y, fi, g4.y));
        mx = fmaxf(mx, gat_e(a4.z, fi, g4.z));
        mx = fmaxf(mx, gat_e(a4.w, fi, g4.w));
    }
    red[r][jt] = mx; __syncthreads();
    if (jt == 0) { float m = red[r][0]; for (int q = 1; q < 8; q++) m = fmaxf(m, red[r][q]); rowm[r] = m; }
    __syncthreads();
    float m = rowm[r], sm = 0.f;
    for (int j = jt * 4; j < N; j += 32) {
        float4 a4 = *(const float4*)(arow + j);
        float4 g4 = *(const float4*)(f2h + j);
        sm += __expf(gat_e(a4.x, fi, g4.x) - m) + __expf(gat_e(a4.y, fi, g4.y) - m)
            + __expf(gat_e(a4.z, fi, g4.z) - m) + __expf(gat_e(a4.w, fi, g4.w) - m);
    }
    red[r][jt] = sm; __syncthreads();
    if (jt == 0) { float s2 = 0.f; for (int q = 0; q < 8; q++) s2 += red[r][q]; rowinv[r] = 1.f / s2; }
    __syncthreads();

    // ---- pass 2: WMMA accumulate over j chunks ----
    const int l16 = lane & 15, hi = lane >> 4;
    const int i = i0 + l16;
    const float fim = f1h[i], rm = rowm[l16], ri = rowinv[l16];
    const float* ar = adj + (size_t)i * N;
    const int ncol = wave * 16 + l16;
    const _Float16* bcol = Wpt + (size_t)ncol * N;
    v8f acc = {};
    for (int j0 = 0; j0 < N; j0 += 32) {
        alignas(16) float av[16], fv[16];
        load_runs(ar,  j0, hi, av);
        load_runs(f2h, j0, hi, fv);
        if (j0 + 32 < N) __builtin_prefetch(ar + j0 + 32);
        HV16 a, b;
        #pragma unroll
        for (int s = 0; s < 16; s++)
            a.v[s] = (_Float16)(__expf(gat_e(av[s], fim, fv[s]) - rm) * ri);
        const uint4* bp = (const uint4*)(bcol + j0 + hi * 16);
        b.u[0] = bp[0]; b.u[1] = bp[1];
        acc = __builtin_amdgcn_wmma_f32_16x16x32_f16(false, a.v, false, b.v, (short)0, acc, false, false);
    }
    #pragma unroll
    for (int v = 0; v < 8; v++) {
        int row = i0 + v + (hi << 3);
        float val = acc[v];
        val = val > 0.f ? val : (__expf(val) - 1.f); // elu
        feat[(size_t)row * 256 + h * 64 + ncol] = val;
    }
}

// ---------------------------------------------------------------------------
// Single-head masked attention (d=256), tail rows only. Block=256 (8 waves),
// each wave owns 32 output cols. Wht is [256][N] K-major.
// ---------------------------------------------------------------------------
__global__ void __launch_bounds__(256)
gat_sg_attn(const float* __restrict__ adj, const float* __restrict__ f1,
            const float* __restrict__ f2, const _Float16* __restrict__ Wht,
            float* __restrict__ g, int N, int row_base)
{
    const int i0 = row_base + blockIdx.x * 16;
    const int tid = threadIdx.x, wave = tid >> 5, lane = tid & 31;

    __shared__ float red[16][16];
    __shared__ float rowm[16], rowinv[16];

    int r = tid >> 4, jt = tid & 15;
    float fi = f1[i0 + r];
    const float* arow = adj + (size_t)(i0 + r) * N;
    float mx = NEG_INF_F;
    for (int j = jt * 4; j < N; j += 64) {
        float4 a4 = *(const float4*)(arow + j);
        float4 g4 = *(const float4*)(f2 + j);
        mx = fmaxf(mx, gat_e(a4.x, fi, g4.x));
        mx = fmaxf(mx, gat_e(a4.y, fi, g4.y));
        mx = fmaxf(mx, gat_e(a4.z, fi, g4.z));
        mx = fmaxf(mx, gat_e(a4.w, fi, g4.w));
    }
    red[r][jt] = mx; __syncthreads();
    if (jt == 0) { float m = red[r][0]; for (int q = 1; q < 16; q++) m = fmaxf(m, red[r][q]); rowm[r] = m; }
    __syncthreads();
    float m = rowm[r], sm = 0.f;
    for (int j = jt * 4; j < N; j += 64) {
        float4 a4 = *(const float4*)(arow + j);
        float4 g4 = *(const float4*)(f2 + j);
        sm += __expf(gat_e(a4.x, fi, g4.x) - m) + __expf(gat_e(a4.y, fi, g4.y) - m)
            + __expf(gat_e(a4.z, fi, g4.z) - m) + __expf(gat_e(a4.w, fi, g4.w) - m);
    }
    red[r][jt] = sm; __syncthreads();
    if (jt == 0) { float s2 = 0.f; for (int q = 0; q < 16; q++) s2 += red[r][q]; rowinv[r] = 1.f / s2; }
    __syncthreads();

    const int l16 = lane & 15, hi = lane >> 4;
    const int i = i0 + l16;
    const float fim = f1[i], rm = rowm[l16], ri = rowinv[l16];
    const float* ar = adj + (size_t)i * N;
    const int n0 = wave * 32;
    const _Float16* bcol0 = Wht + (size_t)(n0 + l16) * N;
    const _Float16* bcol1 = Wht + (size_t)(n0 + 16 + l16) * N;
    v8f acc0 = {}, acc1 = {};
    for (int j0 = 0; j0 < N; j0 += 32) {
        alignas(16) float av[16], fv[16];
        load_runs(ar, j0, hi, av);
        load_runs(f2, j0, hi, fv);
        if (j0 + 32 < N) __builtin_prefetch(ar + j0 + 32);
        HV16 a, b0, b1;
        #pragma unroll
        for (int s = 0; s < 16; s++)
            a.v[s] = (_Float16)(__expf(gat_e(av[s], fim, fv[s]) - rm) * ri);
        const uint4* bp0 = (const uint4*)(bcol0 + j0 + hi * 16);
        const uint4* bp1 = (const uint4*)(bcol1 + j0 + hi * 16);
        b0.u[0] = bp0[0]; b0.u[1] = bp0[1];
        b1.u[0] = bp1[0]; b1.u[1] = bp1[1];
        acc0 = __builtin_amdgcn_wmma_f32_16x16x32_f16(false, a.v, false, b0.v, (short)0, acc0, false, false);
        acc1 = __builtin_amdgcn_wmma_f32_16x16x32_f16(false, a.v, false, b1.v, (short)0, acc1, false, false);
    }
    #pragma unroll
    for (int v = 0; v < 8; v++) {
        int row = blockIdx.x * 16 + v + (hi << 3); // local row in [0,1536)
        g[(size_t)row * 256 + n0 + l16]      = acc0[v];
        g[(size_t)row * 256 + n0 + l16 + 16] = acc1[v];
    }
}

// ---------------------------------------------------------------------------
// Bipartite column softmax: softmax(adj[ds*1536:(ds+1)*1536, 3072:] * 20, axis=0)
// ---------------------------------------------------------------------------
__global__ void __launch_bounds__(256)
bi_softmax(const float* __restrict__ adj, float* __restrict__ out, int N)
{
    const int ds = blockIdx.y, c = blockIdx.x;
    const float* base = adj + (size_t)(ds * 1536) * N + 3072 + c;
    float* ob = out + (size_t)ds * 1536 * 1536;
    __shared__ float red[256];
    int tid = threadIdx.x;
    float mx = -3.4e38f;
    for (int r = tid; r < 1536; r += 256) mx = fmaxf(mx, base[(size_t)r * N]);
    red[tid] = mx; __syncthreads();
    for (int s = 128; s > 0; s >>= 1) { if (tid < s) red[tid] = fmaxf(red[tid], red[tid + s]); __syncthreads(); }
    mx = red[0]; __syncthreads();
    float sm = 0.f;
    for (int r = tid; r < 1536; r += 256) sm += __expf((base[(size_t)r * N] - mx) * 20.f);
    red[tid] = sm; __syncthreads();
    for (int s = 128; s > 0; s >>= 1) { if (tid < s) red[tid] += red[tid + s]; __syncthreads(); }
    float inv = 1.f / red[0];
    for (int r = tid; r < 1536; r += 256)
        ob[(size_t)r * 1536 + c] = __expf((base[(size_t)r * N] - mx) * 20.f) * inv;
}

// ---------------------------------------------------------------------------
extern "C" void kernel_launch(void* const* d_in, const int* in_sizes, int n_in,
                              void* d_out, int out_size, void* d_ws, size_t ws_size,
                              hipStream_t stream)
{
    const int N = 4608, F = 512, FO = 256, FA = 64, H = 4, D = 64, ML = 512, NT = 1536;

    const float* x   = (const float*)d_in[0];
    const float* Wb  = (const float*)d_in[1];
    const float* bb  = (const float*)d_in[2];
    const float* Wa  = (const float*)d_in[3];
    const float* ba  = (const float*)d_in[4];
    const float* Whd = (const float*)d_in[5]; // (4,256,64)
    const float* ah  = (const float*)d_in[6]; // (4,128)
    const float* Wo  = (const float*)d_in[7]; // (256,256)
    const float* ao  = (const float*)d_in[8]; // (512,)
    const float* Wl  = (const float*)d_in[9]; // (256,512)
    const float* bl  = (const float*)d_in[10];
    float* out = (float*)d_out;

    char* ws = (char*)d_ws;
    size_t off = 0;
    auto alloc = [&](size_t bytes) -> void* {
        off = (off + 255) & ~(size_t)255;
        void* p = ws + off;
        off += bytes;
        return p;
    };

    _Float16* xh    = (_Float16*)alloc((size_t)N * F * 2);
    _Float16* WbT   = (_Float16*)alloc((size_t)F * FO * 2);   // [256][512]
    float*    x1    = (float*)   alloc((size_t)N * FO * 4);
    _Float16* x1h   = (_Float16*)alloc((size_t)N * FO * 2);
    _Float16* WaT   = (_Float16*)alloc((size_t)FO * FA * 2);  // [64][256]
    float*    af    = (float*)   alloc((size_t)N * FA * 4);
    float*    naf   = (float*)   alloc((size_t)N * FA * 4);
    _Float16* nafh  = (_Float16*)alloc((size_t)N * FA * 2);   // already K-major panel
    float*    scol  = (float*)   alloc((size_t)FA * 4);
    float*    rinv  = (float*)   alloc((size_t)N * 4);
    float*    adj   = (float*)   alloc((size_t)N * N * 4);
    float*    x1r   = (float*)   alloc((size_t)N * FO * 4);
    _Float16* x1rh  = (_Float16*)alloc((size_t)N * FO * 2);
    _Float16* WhdT  = (_Float16*)alloc((size_t)H * FO * D * 2); // [h][64][256]
    float*    Whm   = (float*)   alloc((size_t)H * N * D * 4);  // [h][N][64]
    _Float16* WhmT  = (_Float16*)alloc((size_t)H * N * D * 2);  // [h][64][N]
    float*    f1m   = (float*)   alloc((size_t)H * N * 4);
    float*    f2m   = (float*)   alloc((size_t)H * N * 4);
    float*    feat  = (float*)   alloc((size_t)N * 256 * 4);
    float*    x2    = (float*)   alloc((size_t)N * 256 * 4);
    _Float16* x2h   = (_Float16*)alloc((size_t)N * 256 * 2);
    _Float16* WoT   = (_Float16*)alloc((size_t)256 * 256 * 2);
    float*    WhO   = (float*)   alloc((size_t)N * 256 * 4);
    _Float16* WhOT  = (_Float16*)alloc((size_t)N * 256 * 2);  // [256][N]
    float*    f1s   = (float*)   alloc((size_t)N * 4);
    float*    f2s   = (float*)   alloc((size_t)N * 4);
    float*    g     = (float*)   alloc((size_t)NT * 256 * 4);
    _Float16* x3h   = (_Float16*)alloc((size_t)NT * 256 * 2);
    _Float16* WlT   = (_Float16*)alloc((size_t)256 * ML * 2); // [512][256]

    auto cgrid = [](size_t n) { return dim3((unsigned)((n + 255) / 256)); };

    // ---- x1 = x @ W_before + b_before ----
    conv_f16<<<cgrid((size_t)N * F), 256, 0, stream>>>(x, xh, (size_t)N * F);
    tconv_f16<<<cgrid((size_t)F * FO), 256, 0, stream>>>(Wb, WbT, F, FO, (size_t)F * FO);
    wmma_gemm<<<dim3(FO / 64, N / 64, 1), 256, 0, stream>>>(
        xh, WbT, x1, N, FO, F, F, F, FO, bb, nullptr, 0, 0, 0, 0);

    // ---- adjacency ----
    conv_f16<<<cgrid((size_t)N * FO), 256, 0, stream>>>(x1, x1h, (size_t)N * FO);
    tconv_f16<<<cgrid((size_t)FO * FA), 256, 0, stream>>>(Wa, WaT, FO, FA, (size_t)FO * FA);
    wmma_gemm<<<dim3(1, N / 64, 1), 256, 0, stream>>>(
        x1h, WaT, af, N, FA, FO, FO, FO, FA, ba, nullptr, 0, 0, 0, 0);
    relu_conv<<<cgrid((size_t)N * FO), 256, 0, stream>>>(x1, x1r, x1rh, (size_t)N * FO);
    row_normalize<<<cgrid(N), 256, 0, stream>>>(af, naf, nafh, N);
    colsum64<<<64, 256, 0, stream>>>(naf, scol, N);
    rinv_kernel<<<cgrid(N), 256, 0, stream>>>(naf, scol, rinv, N);
    wmma_gemm<<<dim3(N / 64, N / 64, 1), 256, 0, stream>>>(
        nafh, nafh, adj, N, N, FA, FA, FA, N, nullptr, rinv, 0, 0, 0, 0);

    // ---- multi-head GAT ----
    tconv_f16<<<cgrid((size_t)H * FO * D), 256, 0, stream>>>(Whd, WhdT, FO, D, (size_t)H * FO * D);
    wmma_gemm<<<dim3(1, N / 64, H), 256, 0, stream>>>(
        x1rh, WhdT, Whm, N, D, FO, FO, FO, D, nullptr, nullptr, 0,
        0, (long)D * FO, (long)N * D);
    tconv_f16<<<cgrid((size_t)H * N * D), 256, 0, stream>>>(Whm, WhmT, N, D, (size_t)H * N * D);
    f12_mh<<<cgrid((size_t)H * N), 256, 0, stream>>>(Whm, ah, f1m, f2m, N);
    gat_mh_attn<<<dim3(N / 16, H), 128, 0, stream>>>(adj, f1m, f2m, WhmT, feat, N);

    // ---- x2 = feat + relu(x1); single-head GAT (tail rows only) ----
    add_conv<<<cgrid((size_t)N * 256), 256, 0, stream>>>(feat, x1r, x2, x2h, (size_t)N * 256);
    tconv_f16<<<cgrid((size_t)256 * 256), 256, 0, stream>>>(Wo, WoT, 256, 256, (size_t)256 * 256);
    wmma_gemm<<<dim3(256 / 64, N / 64, 1), 256, 0, stream>>>(
        x2h, WoT, WhO, N, 256, 256, 256, 256, 256, nullptr, nullptr, 0, 0, 0, 0);
    tconv_f16<<<cgrid((size_t)N * 256), 256, 0, stream>>>(WhO, WhOT, N, 256, (size_t)N * 256);
    f12_sg<<<cgrid(N), 256, 0, stream>>>(WhO, ao, f1s, f2s, N);
    gat_sg_attn<<<dim3(NT / 16), 256, 0, stream>>>(adj, f1s, f2s, WhOT, g, N, 3072);

    // ---- x3 = elu(g + x2_tail); out0 = x3 @ W_lin1 + b_lin1 ----
    x3_elu<<<cgrid((size_t)NT * 256), 256, 0, stream>>>(g, x2, x3h, (size_t)NT * 256);
    tconv_f16<<<cgrid((size_t)256 * ML), 256, 0, stream>>>(Wl, WlT, 256, ML, (size_t)256 * ML);
    wmma_gemm<<<dim3(ML / 64, NT / 64, 1), 256, 0, stream>>>(
        x3h, WlT, out, NT, ML, 256, 256, 256, ML, bl, nullptr, 0, 0, 0, 0);

    // ---- bipartite softmaxes ----
    bi_softmax<<<dim3(1536, 2), 256, 0, stream>>>(adj, out + (size_t)NT * ML, N);
}
